// AttentionBlock_31482110280279
// MI455X (gfx1250) — matrix operand order repaired and checked by reading, not compiled, not run on hardware
//
#include <hip/hip_runtime.h>
#include <hip/hip_bf16.h>

// Problem dims
#define BB   4
#define CC   1024
#define SS   4096
#define KK   1024   // == V dim

typedef __bf16 v8bf  __attribute__((ext_vector_type(8)));
typedef __bf16 v16bf __attribute__((ext_vector_type(16)));
typedef float  v8f   __attribute__((ext_vector_type(8)));
typedef unsigned int u32x4 __attribute__((ext_vector_type(4)));
typedef int          i32x4 __attribute__((ext_vector_type(4)));
typedef int          i32x8 __attribute__((ext_vector_type(8)));

// therock-10.0 HIP headers ship the TDM descriptor header -> 6-arg builtin;
// ROCm 7.2 does not -> 5-arg builtin.
#if __has_include(<hip/amd_detail/amd_gfx1250_TDM.h>)
#define TDM_6ARG 1
#endif

// Build a 16-element bf16 fragment from two aligned 8-element LDS chunks.
__device__ __forceinline__ v16bf ld2(const __bf16* p0, const __bf16* p1) {
    v8bf lo = *(const v8bf*)p0;
    v8bf hi = *(const v8bf*)p1;
    v16bf r;
#pragma unroll
    for (int e = 0; e < 8; ++e) { r[e] = lo[e]; r[8 + e] = hi[e]; }
    return r;
}

// Issue one async global->LDS 128b copy (per active lane). ASYNCcnt-tracked.
__device__ __forceinline__ void async_copy_b128(unsigned lds_byte_off,
                                                const void* gaddr) {
    asm volatile("global_load_async_to_lds_b128 %0, %1, off"
                 :: "v"(lds_byte_off), "v"(gaddr) : "memory");
}
__device__ __forceinline__ void wait_asynccnt0() {
    asm volatile("s_wait_asynccnt 0x0" ::: "memory");
}

// TDM: DMA a dense 2D tile (rows x rowlen_el of 2-byte elements, row stride
// = rowlen_el) from global memory into LDS at lds_byte_off. TENSORcnt-tracked.
// D# layout per CDNA5 ISA ch.10.8 (group0 128b, group1 256b, groups2/3 zero).
__device__ __forceinline__ void tdm_load_2d_bf16(unsigned lds_byte_off,
                                                 const void* gptr,
                                                 unsigned rows,
                                                 unsigned rowlen_el) {
    unsigned long long ga = (unsigned long long)(size_t)gptr;
    u32x4 g0;
    g0[0] = 1u;                                   // count=1, user mode
    g0[1] = lds_byte_off;                         // lds_addr
    g0[2] = (unsigned)(ga & 0xFFFFFFFFu);         // global_addr[31:0]
    g0[3] = (unsigned)((ga >> 32) & 0x01FFFFFFu)  // global_addr[56:32]
            | (2u << 30);                         // type=2 ("image")
    i32x8 g1;
    g1[0] = (int)(1u << 16);                      // data_size=1 (2B), mask=0
    g1[1] = (int)((rowlen_el & 0xFFFFu) << 16);   // tensor_dim0[15:0] @bit48
    g1[2] = (int)(((rowlen_el >> 16) & 0xFFFFu)   // tensor_dim0[31:16]
            | ((rows & 0xFFFFu) << 16));          // tensor_dim1[15:0]
    g1[3] = (int)(((rows >> 16) & 0xFFFFu)        // tensor_dim1[31:16]
            | ((rowlen_el & 0xFFFFu) << 16));     // tile_dim0
    g1[4] = (int)(rows & 0xFFFFu);                // tile_dim1 (tile_dim2=0)
    g1[5] = (int)rowlen_el;                       // tensor_dim0_stride[31:0]
    g1[6] = 0;                                    // stride hi / dim1_stride lo
    g1[7] = 0;
    i32x4 z4; z4[0] = z4[1] = z4[2] = z4[3] = 0;
#ifdef TDM_6ARG
    i32x8 z8;
#pragma unroll
    for (int e = 0; e < 8; ++e) z8[e] = 0;
    __builtin_amdgcn_tensor_load_to_lds(g0, g1, z4, z4, z8, 0);
#else
    __builtin_amdgcn_tensor_load_to_lds(g0, g1, z4, z4, 0);
#endif
}

// ---------------------------------------------------------------------------
// Kernel 0: copy input [B,C,S] into top half of out [B, C+V, S]
// ---------------------------------------------------------------------------
__global__ __launch_bounds__(256) void copy_in(const float* __restrict__ in,
                                               float* __restrict__ out) {
    size_t i = (size_t)blockIdx.x * blockDim.x + threadIdx.x; // float4 index
    const size_t CS4 = (size_t)CC * SS / 4;
    size_t b = i / CS4;
    size_t r = i - b * CS4;
    float4 v = ((const float4*)in)[i];
    ((float4*)out)[b * ((size_t)2 * CC * SS / 4) + r] = v;
}

// ---------------------------------------------------------------------------
// Kernel 1: projection GEMM:  out[b,s,n] = (x[b,s,:] @ W + bias) * scale
//   x[b,s,c] = input[b,c,s] (read coalesced along s, transposed into LDS)
//   Tiles: 128(s) x 128(n), BK = 32. 8 waves, each wave 32x64.
// ---------------------------------------------------------------------------
__global__ __launch_bounds__(256) void qkv_gemm(const float* __restrict__ x,
                                                const float* __restrict__ W,
                                                const float* __restrict__ bias,
                                                __bf16* __restrict__ outp,
                                                float scale) {
    __shared__ __bf16 sA[128 * 40];  // [s_local][c_local] bf16, pad to 40
    __shared__ __bf16 sB[128 * 40];  // [n_local][c_local] bf16, pad to 40

    const int b  = blockIdx.z;
    const int s0 = blockIdx.x * 128;
    const int n0 = blockIdx.y * 128;
    const int tid  = threadIdx.x;
    const int lane = tid & 31;
    const int w    = tid >> 5;
    const int rm   = (w & 3) * 32;
    const int cn   = (w >> 2) * 64;

    v8f acc[2][4];
#pragma unroll
    for (int i = 0; i < 2; ++i)
#pragma unroll
        for (int j = 0; j < 4; ++j)
#pragma unroll
            for (int e = 0; e < 8; ++e) acc[i][j][e] = 0.f;

    const int cl = tid >> 3;
    const int sg = (tid & 7) * 16;

    for (int c0 = 0; c0 < CC; c0 += 32) {
        const float* gA = x + ((size_t)b * CC + (c0 + cl)) * SS + s0 + sg;
#pragma unroll
        for (int u = 0; u < 4; ++u) {
            float4 v = ((const float4*)gA)[u];
            int sl = sg + u * 4;
            sA[(sl + 0) * 40 + cl] = (__bf16)v.x;
            sA[(sl + 1) * 40 + cl] = (__bf16)v.y;
            sA[(sl + 2) * 40 + cl] = (__bf16)v.z;
            sA[(sl + 3) * 40 + cl] = (__bf16)v.w;
        }
        const float* gB = W + (size_t)(c0 + cl) * KK + n0 + sg;
#pragma unroll
        for (int u = 0; u < 4; ++u) {
            float4 v = ((const float4*)gB)[u];
            int nl = sg + u * 4;
            sB[(nl + 0) * 40 + cl] = (__bf16)v.x;
            sB[(nl + 1) * 40 + cl] = (__bf16)v.y;
            sB[(nl + 2) * 40 + cl] = (__bf16)v.z;
            sB[(nl + 3) * 40 + cl] = (__bf16)v.w;
        }
        __syncthreads();

        v16bf af[2];
#pragma unroll
        for (int i = 0; i < 2; ++i) {
            int row = rm + i * 16 + (lane & 15);
            int kb  = (lane >> 4) * 8;
            af[i] = ld2(&sA[row * 40 + kb], &sA[row * 40 + kb + 16]);
        }
#pragma unroll
        for (int j = 0; j < 4; ++j) {
            int col = cn + j * 16 + (lane & 15);
            int kb  = (lane >> 4) * 8;
            v16bf bf_ = ld2(&sB[col * 40 + kb], &sB[col * 40 + kb + 16]);
#pragma unroll
            for (int i = 0; i < 2; ++i)
                acc[i][j] = __builtin_amdgcn_wmma_f32_16x16x32_bf16(
                    false, af[i], false, bf_, (short)0, acc[i][j], false, false);
        }
        __syncthreads();
    }

#pragma unroll
    for (int j = 0; j < 4; ++j) {
        int n = n0 + cn + j * 16 + (lane & 15);
        float bv = bias[n];
#pragma unroll
        for (int i = 0; i < 2; ++i) {
#pragma unroll
            for (int r = 0; r < 8; ++r) {
                int m = rm + i * 16 + r + ((lane >> 4) << 3);
                size_t off = ((size_t)b * SS + (s0 + m)) * KK + n;
                outp[off] = (__bf16)((acc[i][j][r] + bv) * scale);
            }
        }
    }
}

// ---------------------------------------------------------------------------
// Kernel 2: flash-style causal attention, head dim 1024.
//   Q tile staged with global_load_async_to_lds_b128 (ASYNCcnt),
//   K chunk staged with the Tensor Data Mover (TENSORcnt),
//   8 waves split the 1024-deep score dot, ds_add_f32 reduction,
//   wave0 online softmax, all waves P@V on a 128-wide output slice,
//   output staged through LDS for coalesced 64B bursts.
// LDS byte map:
//   sQ  @ 0       16*1032 bf16  (33024)
//   sK  @ 33024   32*1024 bf16  (65536)   <- TDM target, unpadded rows
//   sVt @ 98560   1024*40 bf16  (81920)   [vcol][key]
//   sS  @ 180480  512 f32       ( 2048)
//   sP  @ 182528  16*40 bf16    ( 1280)
//   sC  @ 183808  16 f32        (   64)
//   sL  @ 183872  16 f32        (   64)   total 183936
//   sO  @ 0       1024*16 f32   (65536)   reused post-loop over sQ/sK
// ---------------------------------------------------------------------------
#define QSTR 1032
#define SK_OFF 33024u
#define ATTN_LDS 183936

__global__ __launch_bounds__(256) void attn(const __bf16* __restrict__ Qb,
                                            const __bf16* __restrict__ Kb,
                                            const __bf16* __restrict__ Vb,
                                            const float* __restrict__ alibi_p,
                                            float* __restrict__ out) {
    extern __shared__ __align__(16) char smem[];
    __bf16* sQ  = (__bf16*)smem;                   // 16*1032
    __bf16* sK  = (__bf16*)(smem + SK_OFF);        // 32*1024 (stride 1024)
    __bf16* sVt = sK + 32 * 1024;                  // 1024*40  [vcol][key]
    float*  sS  = (float*)(sVt + 1024 * 40);       // 512 = [n][m]
    __bf16* sP  = (__bf16*)(sS + 512);             // 16*40
    float*  sC  = (float*)(sP + 16 * 40);          // 16 correction factors
    float*  sL  = sC + 16;                         // 16 final row sums
    float*  sO  = (float*)smem;                    // 1024*16, post-loop reuse

    const int b    = blockIdx.y;
    const int q0   = blockIdx.x * 16;
    const int tid  = threadIdx.x;
    const int lane = tid & 31;
    const int w    = tid >> 5;  // 0..7

    // Async-stage Q tile (16 x 1024 bf16): 8 x b128 per thread.
    {
        int r  = tid >> 4;           // 0..15
        int c0 = (tid & 15) * 64;
        const __bf16* g = Qb + ((size_t)b * SS + (q0 + r)) * KK + c0;
#pragma unroll
        for (int u = 0; u < 8; ++u)
            async_copy_b128((unsigned)((r * QSTR + c0 + u * 8) * 2), g + u * 8);
        wait_asynccnt0();
    }

    v8f o[8];
#pragma unroll
    for (int j = 0; j < 8; ++j)
#pragma unroll
        for (int e = 0; e < 8; ++e) o[j][e] = 0.f;

    float mrun = -1e30f, lrun = 0.f;  // wave0 lanes 0..15 only
    const float sig = 1.f / (1.f + __expf(-alibi_p[0]));

    const int jmax = (q0 + 15) >> 5;
    for (int jc = 0; jc <= jmax; ++jc) {
        const int t0 = jc * 32;
        __syncthreads();  // protect LDS reuse from previous iteration

        // TDM: DMA K chunk (32 rows x 1024 bf16, dense) into sK.
        if (w == 0)
            tdm_load_2d_bf16(SK_OFF, Kb + ((size_t)b * SS + t0) * KK, 32, KK);

        // V chunk: load rows, write transposed [vcol][key]; zero score tile.
        {
            int kl = tid >> 3;          // key 0..31
            int c0 = (tid & 7) * 128;
            const __bf16* gv = Vb + ((size_t)b * SS + (t0 + kl)) * KK + c0;
#pragma unroll
            for (int u = 0; u < 16; ++u) {
                v8bf v = *(const v8bf*)(gv + u * 8);
#pragma unroll
                for (int e = 0; e < 8; ++e)
                    sVt[(c0 + u * 8 + e) * 40 + kl] = v[e];
            }
            sS[tid] = 0.f;
            sS[tid + 256] = 0.f;
        }
        if (w == 0) __builtin_amdgcn_s_wait_tensorcnt(0);
        __syncthreads();

        // Partial scores over channels [w*128, w*128+128)
        {
            v8f sc[2];
#pragma unroll
            for (int n = 0; n < 2; ++n)
#pragma unroll
                for (int e = 0; e < 8; ++e) sc[n][e] = 0.f;
#pragma unroll
            for (int kk = 0; kk < 4; ++kk) {
                int k0  = w * 128 + kk * 32;
                int row = (lane & 15);
                int kb  = (lane >> 4) * 8;
                v16bf a = ld2(&sQ[row * QSTR + k0 + kb],
                              &sQ[row * QSTR + k0 + kb + 16]);
#pragma unroll
                for (int n = 0; n < 2; ++n) {
                    int key = n * 16 + (lane & 15);
                    int kbb = (lane >> 4) * 8;
                    v16bf bb = ld2(&sK[key * 1024 + k0 + kbb],
                                   &sK[key * 1024 + k0 + kbb + 16]);
                    sc[n] = __builtin_amdgcn_wmma_f32_16x16x32_bf16(
                        false, a, false, bb, (short)0, sc[n], false, false);
                }
            }
#pragma unroll
            for (int n = 0; n < 2; ++n)
#pragma unroll
                for (int r = 0; r < 8; ++r) {
                    int m  = r + ((lane >> 4) << 3);
                    int nn = n * 16 + (lane & 15);
                    atomicAdd(&sS[nn * 16 + m], sc[n][r]);  // ds_add_f32
                }
        }
        __syncthreads();

        // Online softmax: wave0, lane = query row
        if (w == 0 && lane < 16) {
            int m = lane;
            int q = q0 + m;
            float vals[32];
            float rmax = -1e30f;
#pragma unroll
            for (int n = 0; n < 32; ++n) {
                int t = t0 + n;
                float v = sS[n * 16 + m];
                v = (t > q) ? -1e30f : (v - sig * (float)(q - t));
                vals[n] = v;
                rmax = fmaxf(rmax, v);
            }
            float mnew = fmaxf(mrun, rmax);
            float corr = __expf(mrun - mnew);
            float ls = 0.f;
#pragma unroll
            for (int n = 0; n < 32; ++n) {
                float p = __expf(vals[n] - mnew);
                ls += p;
                sP[m * 40 + n] = (__bf16)p;
            }
            lrun = lrun * corr + ls;
            mrun = mnew;
            sC[m] = corr;
        }
        __syncthreads();

        // Rescale O and accumulate P @ V for this wave's 128-col slice
        {
            float cf[8];
#pragma unroll
            for (int r = 0; r < 8; ++r) cf[r] = sC[r + ((lane >> 4) << 3)];
#pragma unroll
            for (int j = 0; j < 8; ++j)
#pragma unroll
                for (int r = 0; r < 8; ++r) o[j][r] *= cf[r];

            int row = lane & 15;
            int kb  = (lane >> 4) * 8;
            v16bf a = ld2(&sP[row * 40 + kb], &sP[row * 40 + kb + 16]);
#pragma unroll
            for (int j = 0; j < 8; ++j) {
                int vcol = w * 128 + j * 16 + (lane & 15);
                int kbb  = (lane >> 4) * 8;
                v16bf bb = ld2(&sVt[vcol * 40 + kbb],
                               &sVt[vcol * 40 + kbb + 16]);
                o[j] = __builtin_amdgcn_wmma_f32_16x16x32_bf16(
                    false, a, false, bb, (short)0, o[j], false, false);
            }
        }
    }

    __syncthreads();
    if (w == 0 && lane < 16) sL[lane] = lrun;
    __syncthreads();

    // Normalize into LDS staging tile sO[n][m] (sQ/sK regions are dead now).
    {
        float inv[8];
#pragma unroll
        for (int r = 0; r < 8; ++r) inv[r] = 1.f / sL[r + ((lane >> 4) << 3)];
#pragma unroll
        for (int j = 0; j < 8; ++j) {
            int n = w * 128 + j * 16 + (lane & 15);
#pragma unroll
            for (int r = 0; r < 8; ++r) {
                int m = r + ((lane >> 4) << 3);
                sO[n * 16 + m] = o[j][r] * inv[r];
            }
        }
    }
    __syncthreads();

    // Coalesced stores: 4 lanes cover one 64B row of 16 s-values.
    {
        int nbase = tid >> 2;          // 0..63
        int qoff  = (tid & 3) * 4;     // 0,4,8,12
#pragma unroll
        for (int it = 0; it < 16; ++it) {
            int n = nbase + it * 64;
            float4 v = *(const float4*)&sO[n * 16 + qoff];
            size_t off = ((size_t)b * (2 * CC) + (CC + n)) * SS + (q0 + qoff);
            *(float4*)(out + off) = v;
        }
    }
}

// ---------------------------------------------------------------------------
extern "C" void kernel_launch(void* const* d_in, const int* in_sizes, int n_in,
                              void* d_out, int out_size, void* d_ws, size_t ws_size,
                              hipStream_t stream) {
    const float* input = (const float*)d_in[0];
    // d_in[1] = frame_no (arange) -> index math used instead
    const float* Wq = (const float*)d_in[2];
    const float* bq = (const float*)d_in[3];
    const float* Wk = (const float*)d_in[4];
    const float* bk = (const float*)d_in[5];
    const float* Wv = (const float*)d_in[6];
    const float* bv = (const float*)d_in[7];
    const float* alibi = (const float*)d_in[8];
    float* out = (float*)d_out;

    // Workspace: bf16 Q | K | V, each B*S*1024 (32 MB) -> 96 MB total
    __bf16* Qb = (__bf16*)d_ws;
    __bf16* Kb = Qb + (size_t)BB * SS * KK;
    __bf16* Vb = Kb + (size_t)BB * SS * KK;

    // 1) input -> out[:, :C, :]
    copy_in<<<dim3((BB * CC * SS / 4) / 256), dim3(256), 0, stream>>>(input, out);

    // 2) QKV projections (1/sqrt(K)=1/32 folded into Q)
    dim3 gg(SS / 128, KK / 128, BB), bb(256);
    qkv_gemm<<<gg, bb, 0, stream>>>(input, Wq, bq, Qb, 0.03125f);
    qkv_gemm<<<gg, bb, 0, stream>>>(input, Wk, bk, Kb, 1.0f);
    qkv_gemm<<<gg, bb, 0, stream>>>(input, Wv, bv, Vb, 1.0f);

    // 3) causal attention + ALiBi -> out[:, C:, :]
    attn<<<dim3(SS / 16, BB), dim3(256), ATTN_LDS, stream>>>(Qb, Kb, Vb, alibi, out);
}
